// TransformerBlock_39556648796516
// MI455X (gfx1250) — compile-verified, matrix operand order checked by reading, not run on hardware
//
#include <hip/hip_runtime.h>

typedef _Float16 f16;
typedef __attribute__((ext_vector_type(16))) _Float16 v16h;
typedef __attribute__((ext_vector_type(8)))  _Float16 v8h;
typedef __attribute__((ext_vector_type(8)))  float    v8f;
typedef __attribute__((ext_vector_type(4)))  unsigned int u32x4;
typedef __attribute__((ext_vector_type(8)))  int i32x8;
typedef __attribute__((ext_vector_type(4)))  int i32x4;

static constexpr int kT   = 2048;
static constexpr int kD   = 1024;
static constexpr int kH   = 16;
static constexpr int kDH  = 64;
static constexpr int kDFF = 4096;

#if defined(__gfx1250__) && __has_builtin(__builtin_amdgcn_tensor_load_to_lds)
#define USE_TDM 1
#else
#define USE_TDM 0
#endif

union HF { v16h v; v8h h[2]; };

// CDNA5 16-bit A/B fragment: lane-group g selects K-half; halves
// e=0..7 -> k = 8g + e, e=8..15 -> k = 16 + 8g + (e-8). Two b128 loads.
__device__ __forceinline__ v16h ld_frag(const f16* p, int g) {
  HF u;
  u.h[0] = *(const v8h*)(p + 8 * g);
  u.h[1] = *(const v8h*)(p + 16 + 8 * g);
  return u.v;
}

__device__ __forceinline__ v8f wmma32(v16h a, v16h b, v8f c) {
  return __builtin_amdgcn_wmma_f32_16x16x32_f16(false, a, false, b,
                                                (short)0, c, false, false);
}

__device__ __forceinline__ float gelu_f(float x) {
  const float c = 0.7978845608028654f; // sqrt(2/pi)
  return 0.5f * x * (1.0f + tanhf(c * (x + 0.044715f * x * x * x)));
}

#if USE_TDM
// TDM 2D tile load: f16 elements, row-major tensor with row stride
// `stride0` elements; tile is tile0 x tile1 elements; LDS destination gets
// 4 DWORDs of padding every 16 DWORDs (=> row stride 40 halves for tile0=32).
// D# bit layout per CDNA5 ISA ch.8 (group0 128b, group1 256b; groups 2/3 zero
// => 2D tensor semantics, tile_dim2 = 0).
__device__ __forceinline__ void tdm_load_2d(unsigned int lds_addr,
                                            const f16* gptr,
                                            unsigned int tile0,
                                            unsigned int tile1,
                                            unsigned int stride0) {
  unsigned long long ga = (unsigned long long)(size_t)(const void*)gptr;
  u32x4 g0;
  g0[0] = 1u;                                   // count=1, user mode
  g0[1] = lds_addr;                             // lds_addr [63:32]
  g0[2] = (unsigned int)ga;                     // global_addr [95:64]
  g0[3] = (unsigned int)((ga >> 32) & 0x1FFFFFFull) | (2u << 30); // type=2
  unsigned int tdim0 = stride0;                 // no OOB by construction
  unsigned int tdim1 = 1u << 20;
  i32x8 g1;
  g1[0] = (int)((1u << 16)                      // data_size = 2 bytes
              | (1u << 20)                      // pad_enable
              | (3u << 22)                      // pad_interval: 16 DWORDs
              | (3u << 25));                    // pad_amount:   4 DWORDs
  g1[1] = (int)((tdim0 & 0xFFFFu) << 16);       // [63:48] tensor_dim0 lo
  g1[2] = (int)((tdim0 >> 16) | ((tdim1 & 0xFFFFu) << 16));
  g1[3] = (int)((tdim1 >> 16) | (tile0 << 16)); // [127:112] tile_dim0
  g1[4] = (int)tile1;                           // [143:128] tile_dim1
  g1[5] = (int)stride0;                         // tensor_dim0_stride lo32
  g1[6] = 0;                                    // stride hi / dim1_stride
  g1[7] = 0;
  i32x4 z4 = {0, 0, 0, 0};
#if __clang_major__ >= 23
  i32x8 z8 = {0, 0, 0, 0, 0, 0, 0, 0};
  __builtin_amdgcn_tensor_load_to_lds(g0, g1, z4, z4, z8, 0);
#else
  __builtin_amdgcn_tensor_load_to_lds(g0, g1, z4, z4, 0);
#endif
}
#endif

// ---------------------------------------------------------------- cvt f32->f16
__global__ __launch_bounds__(256) void cvt_f16(const float* __restrict__ in,
                                               f16* __restrict__ out, int n) {
  int i = blockIdx.x * 256 + threadIdx.x;
  if (i < n) out[i] = (f16)in[i];
}

// ---------------------------------------------------------------- LayerNorm
__global__ __launch_bounds__(256) void ln_f16(const float* __restrict__ x,
                                              const float* __restrict__ w,
                                              const float* __restrict__ b,
                                              f16* __restrict__ out) {
  __shared__ float rs[8], rs2[8];
  size_t base = (size_t)blockIdx.x * kD;
  float s = 0.f, s2 = 0.f;
  for (int i = threadIdx.x; i < kD; i += 256) {
    float v = x[base + i];
    s += v; s2 += v * v;
  }
  #pragma unroll
  for (int m = 16; m; m >>= 1) { s += __shfl_xor(s, m, 32); s2 += __shfl_xor(s2, m, 32); }
  int wv = threadIdx.x >> 5;
  if ((threadIdx.x & 31) == 0) { rs[wv] = s; rs2[wv] = s2; }
  __syncthreads();
  if (wv == 0) {
    float a  = (threadIdx.x < 8) ? rs[threadIdx.x]  : 0.f;
    float a2 = (threadIdx.x < 8) ? rs2[threadIdx.x] : 0.f;
    #pragma unroll
    for (int m = 4; m; m >>= 1) { a += __shfl_xor(a, m, 32); a2 += __shfl_xor(a2, m, 32); }
    if (threadIdx.x == 0) { rs[0] = a; rs2[0] = a2; }
  }
  __syncthreads();
  float mu = rs[0] / kD;
  float var = rs2[0] / kD - mu * mu;
  float rstd = rsqrtf(var + 1e-5f);
  for (int i = threadIdx.x; i < kD; i += 256)
    out[base + i] = (f16)((x[base + i] - mu) * rstd * w[i] + b[i]);
}

// ---------------------------------------------------------------- GEMM
// C[M,N] = A[M,K](f16,row) * W[N,K](f16,row)^T, fp32 accumulate via WMMA.
// Block tile 128x64, 8 waves of 32x32. K staged in steps of 32 through LDS —
// via double-buffered TDM (tensor_load_to_lds + s_wait_tensorcnt) when
// available, else manual b128 staging.
enum { EP_KQV = 0, EP_ATTNO = 1, EP_UP = 2, EP_DOWN = 3 };

template <int EPI>
__global__ __launch_bounds__(256) void gemm_f16(
    const f16* __restrict__ A, const f16* __restrict__ W,
    const float* __restrict__ bias, const float* __restrict__ resid,
    f16* __restrict__ outk, f16* __restrict__ outq, f16* __restrict__ outv,
    f16* __restrict__ outh, float* __restrict__ outf, int N, int K) {
#if USE_TDM
  __shared__ __align__(16) f16 Al[2][128 * 40];
  __shared__ __align__(16) f16 Bl[2][64 * 40];
#else
  __shared__ __align__(16) f16 Al[1][128 * 40];
  __shared__ __align__(16) f16 Bl[1][64 * 40];
#endif
  const int tid  = threadIdx.x;
  const int wave = tid >> 5, lane = tid & 31;
  const int ln = lane & 15, g = lane >> 4;
  const int m0 = blockIdx.y * 128, n0 = blockIdx.x * 64;
  const int wm = (wave >> 1) * 32, wn = (wave & 1) * 32;

  const v8f vzero = {0, 0, 0, 0, 0, 0, 0, 0};
  v8f acc[2][2];
  #pragma unroll
  for (int mi = 0; mi < 2; ++mi)
    #pragma unroll
    for (int ni = 0; ni < 2; ++ni) acc[mi][ni] = vzero;

#if USE_TDM
  const unsigned int aBytes = 128 * 40 * 2, bBytes = 64 * 40 * 2;
  const unsigned int aLds = (unsigned int)(size_t)(const void*)&Al[0][0];
  const unsigned int bLds = (unsigned int)(size_t)(const void*)&Bl[0][0];
  const int nIter = K >> 5;
  if (tid < 32) {  // one wave drives the tensor DMA
    tdm_load_2d(aLds, A + (size_t)m0 * K, 32, 128, (unsigned int)K);
    tdm_load_2d(bLds, W + (size_t)n0 * K, 32, 64, (unsigned int)K);
  }
  for (int it = 0; it < nIter; ++it) {
    const int p = it & 1;
    if (it + 1 < nIter) {
      if (tid < 32) {
        tdm_load_2d(aLds + (unsigned int)(p ^ 1) * aBytes,
                    A + (size_t)m0 * K + (it + 1) * 32, 32, 128, (unsigned int)K);
        tdm_load_2d(bLds + (unsigned int)(p ^ 1) * bBytes,
                    W + (size_t)n0 * K + (it + 1) * 32, 32, 64, (unsigned int)K);
        // 4 outstanding; wait until only the 2 just-issued remain (in-order)
        __builtin_amdgcn_s_wait_tensorcnt((short)2);
      }
    } else {
      if (tid < 32) __builtin_amdgcn_s_wait_tensorcnt((short)0);
    }
    __syncthreads();
    v16h af[2], bf[2];
    #pragma unroll
    for (int mi = 0; mi < 2; ++mi) af[mi] = ld_frag(&Al[p][(wm + mi * 16 + ln) * 40], g);
    #pragma unroll
    for (int ni = 0; ni < 2; ++ni) bf[ni] = ld_frag(&Bl[p][(wn + ni * 16 + ln) * 40], g);
    #pragma unroll
    for (int mi = 0; mi < 2; ++mi)
      #pragma unroll
      for (int ni = 0; ni < 2; ++ni) acc[mi][ni] = wmma32(af[mi], bf[ni], acc[mi][ni]);
    __syncthreads();
  }
#else
  for (int k0 = 0; k0 < K; k0 += 32) {
    #pragma unroll
    for (int i = 0; i < 2; ++i) {
      int c = tid + 256 * i;
      int r = c >> 2, cc = (c & 3) << 3;
      *(v8h*)&Al[0][r * 40 + cc] = *(const v8h*)&A[(size_t)(m0 + r) * K + k0 + cc];
    }
    {
      int r = tid >> 2, cc = (tid & 3) << 3;
      *(v8h*)&Bl[0][r * 40 + cc] = *(const v8h*)&W[(size_t)(n0 + r) * K + k0 + cc];
    }
    __syncthreads();
    v16h af[2], bf[2];
    #pragma unroll
    for (int mi = 0; mi < 2; ++mi) af[mi] = ld_frag(&Al[0][(wm + mi * 16 + ln) * 40], g);
    #pragma unroll
    for (int ni = 0; ni < 2; ++ni) bf[ni] = ld_frag(&Bl[0][(wn + ni * 16 + ln) * 40], g);
    #pragma unroll
    for (int mi = 0; mi < 2; ++mi)
      #pragma unroll
      for (int ni = 0; ni < 2; ++ni) acc[mi][ni] = wmma32(af[mi], bf[ni], acc[mi][ni]);
    __syncthreads();
  }
#endif

  // epilogue: C layout -> row m = r + 8g, col n = lane%16
  #pragma unroll
  for (int mi = 0; mi < 2; ++mi)
    #pragma unroll
    for (int ni = 0; ni < 2; ++ni)
      #pragma unroll
      for (int r = 0; r < 8; ++r) {
        int t = m0 + wm + mi * 16 + r + 8 * g;
        int j = n0 + wn + ni * 16 + ln;
        float val = acc[mi][ni][r] + bias[j];
        if constexpr (EPI == EP_KQV) {
          int sec = j >> 10;        // 0:k 1:q 2:v  (D = 1024)
          int c   = j & 1023;
          int hh  = c & 15, dd = c >> 4;   // head index is FAST axis
          f16 hv = (f16)val;
          if (sec == 0)       outk[((size_t)hh * kT + t) * kDH + dd] = hv;
          else if (sec == 1)  outq[((size_t)hh * kT + t) * kDH + dd] = hv;
          else                outv[((size_t)hh * kDH + dd) * kT + t] = hv; // V^T
        } else if constexpr (EPI == EP_ATTNO) {
          outf[(size_t)t * N + j] = resid[(size_t)t * N + j] + val;
        } else if constexpr (EPI == EP_UP) {
          outh[(size_t)t * N + j] = (f16)gelu_f(val);
        } else { // EP_DOWN
          outf[(size_t)t * N + j] = resid[(size_t)t * N + j] + val;
        }
      }
}

// ---------------------------------------------------------------- attention
__global__ __launch_bounds__(128) void attn_fa(const f16* __restrict__ qh,
                                               const f16* __restrict__ kh,
                                               const f16* __restrict__ vT,
                                               f16* __restrict__ attn) {
  __shared__ __align__(16) f16 Pl[4][16 * 40];
  const int hh = blockIdx.y;
  const int wave = threadIdx.x >> 5, lane = threadIdx.x & 31;
  const int ln = lane & 15, g = lane >> 4;
  const int q0 = blockIdx.x * 64 + wave * 16;
  const float NEG_INF = -__builtin_inff();

  const f16* qrow = qh + ((size_t)hh * kT + q0 + ln) * kDH;
  v16h qf0 = ld_frag(qrow, g);       // dh 0..31
  v16h qf1 = ld_frag(qrow + 32, g);  // dh 32..63

  const v8f vzero = {0, 0, 0, 0, 0, 0, 0, 0};
  v8f o[4];
  float mrow[8], lrow[8];
  #pragma unroll
  for (int i = 0; i < 4; ++i) o[i] = vzero;
  #pragma unroll
  for (int r = 0; r < 8; ++r) { mrow[r] = NEG_INF; lrow[r] = 0.f; }

  f16* pme = &Pl[wave][0];
  const int jend = q0 + 15;

  for (int j0 = 0; j0 <= jend; j0 += 32) {  // wave-uniform causal bound
    if (j0 + 32 <= jend) {                  // prefetch next K/V tiles
      __builtin_prefetch(kh + ((size_t)hh * kT + j0 + 32 + ln) * kDH, 0, 1);
      __builtin_prefetch(vT + ((size_t)hh * kDH + ln) * kT + j0 + 32, 0, 1);
    }
    v8f s[2]; s[0] = vzero; s[1] = vzero;
    #pragma unroll
    for (int ni = 0; ni < 2; ++ni) {
      const f16* krow = kh + ((size_t)hh * kT + j0 + ni * 16 + ln) * kDH;
      v16h b0 = ld_frag(krow, g);
      v16h b1 = ld_frag(krow + 32, g);
      s[ni] = wmma32(qf0, b0, s[ni]);
      s[ni] = wmma32(qf1, b1, s[ni]);
    }
    #pragma unroll
    for (int r = 0; r < 8; ++r) {
      int qi = q0 + r + 8 * g;
      float s0 = (j0 + ln      <= qi) ? s[0][r] * 8.0f : NEG_INF; // *sqrt(DH)
      float s1 = (j0 + 16 + ln <= qi) ? s[1][r] * 8.0f : NEG_INF;
      float rm = fmaxf(s0, s1);
      #pragma unroll
      for (int m = 8; m; m >>= 1) rm = fmaxf(rm, __shfl_xor(rm, m, 32));
      float mnew  = fmaxf(mrow[r], rm);
      float alpha = expf(mrow[r] - mnew);
      float p0 = expf(s0 - mnew);
      float p1 = expf(s1 - mnew);
      pme[(r + 8 * g) * 40 + ln]      = (f16)p0;
      pme[(r + 8 * g) * 40 + 16 + ln] = (f16)p1;
      float ps = p0 + p1;
      #pragma unroll
      for (int m = 8; m; m >>= 1) ps += __shfl_xor(ps, m, 32);
      lrow[r] = lrow[r] * alpha + ps;
      mrow[r] = mnew;
      #pragma unroll
      for (int sb = 0; sb < 4; ++sb) o[sb][r] *= alpha;
    }
    asm volatile("s_wait_dscnt 0" ::: "memory");  // P tile visible in LDS
    v16h pf = ld_frag(pme + ln * 40, g);          // P as A-matrix 16x32
    #pragma unroll
    for (int sb = 0; sb < 4; ++sb) {
      const f16* vrow = vT + ((size_t)hh * kDH + sb * 16 + ln) * kT + j0;
      v16h vf = ld_frag(vrow, g);                 // contiguous keys (V^T)
      o[sb] = wmma32(pf, vf, o[sb]);
    }
  }

  #pragma unroll
  for (int r = 0; r < 8; ++r) {
    float inv = 1.0f / lrow[r];
    int t = q0 + r + 8 * g;
    #pragma unroll
    for (int sb = 0; sb < 4; ++sb)  // merge: head is SLOW axis
      attn[(size_t)t * kD + hh * kDH + sb * 16 + ln] = (f16)(o[sb][r] * inv);
  }
}

// ---------------------------------------------------------------- launch
extern "C" void kernel_launch(void* const* d_in, const int* in_sizes, int n_in,
                              void* d_out, int out_size, void* d_ws, size_t ws_size,
                              hipStream_t stream) {
  const float* x      = (const float*)d_in[0];
  const float* w_kqv  = (const float*)d_in[1];
  const float* b_kqv  = (const float*)d_in[2];
  const float* w_o    = (const float*)d_in[3];
  const float* b_o    = (const float*)d_in[4];
  const float* ln1w   = (const float*)d_in[5];
  const float* ln1b   = (const float*)d_in[6];
  const float* ln2w   = (const float*)d_in[7];
  const float* ln2b   = (const float*)d_in[8];
  const float* w_up   = (const float*)d_in[9];
  const float* b_up   = (const float*)d_in[10];
  const float* w_down = (const float*)d_in[11];
  const float* b_down = (const float*)d_in[12];

  char* ws = (char*)d_ws;
  size_t off = 0;
  f16* h16  = (f16*)(ws + off); off += (size_t)kT * kD * 2;        // h / h2
  f16* wk16 = (f16*)(ws + off); off += (size_t)3 * kD * kD * 2;
  f16* attn16 = wk16;  // reuse after KQV GEMM (4MB <= 6MB)
  f16* wo16 = (f16*)(ws + off); off += (size_t)kD * kD * 2;
  f16* wu16 = (f16*)(ws + off); off += (size_t)kDFF * kD * 2;
  f16* wd16 = (f16*)(ws + off); off += (size_t)kD * kDFF * 2;
  f16* khd  = (f16*)(ws + off); off += (size_t)kT * kD * 2;        // [H][T][DH]
  f16* qhd  = (f16*)(ws + off); off += (size_t)kT * kD * 2;        // [H][T][DH]
  f16* vTd  = (f16*)(ws + off); off += (size_t)kT * kD * 2;        // [H][DH][T]
  float* x1 = (float*)(ws + off); off += (size_t)kT * kD * 4;
  f16* up16 = (f16*)(ws + off); off += (size_t)kT * kDFF * 2;

  cvt_f16<<<(3 * kD * kD + 255) / 256, 256, 0, stream>>>(w_kqv, wk16, 3 * kD * kD);
  cvt_f16<<<(kD * kD + 255) / 256, 256, 0, stream>>>(w_o, wo16, kD * kD);
  cvt_f16<<<(kDFF * kD + 255) / 256, 256, 0, stream>>>(w_up, wu16, kDFF * kD);
  cvt_f16<<<(kD * kDFF + 255) / 256, 256, 0, stream>>>(w_down, wd16, kD * kDFF);

  ln_f16<<<kT, 256, 0, stream>>>(x, ln1w, ln1b, h16);

  gemm_f16<EP_KQV><<<dim3(3 * kD / 64, kT / 128), 256, 0, stream>>>(
      h16, wk16, b_kqv, nullptr, khd, qhd, vTd, nullptr, nullptr, 3 * kD, kD);

  attn_fa<<<dim3(kT / 64, kH), 128, 0, stream>>>(qhd, khd, vTd, attn16);

  gemm_f16<EP_ATTNO><<<dim3(kD / 64, kT / 128), 256, 0, stream>>>(
      attn16, wo16, b_o, x, nullptr, nullptr, nullptr, nullptr, x1, kD, kD);

  ln_f16<<<kT, 256, 0, stream>>>(x1, ln2w, ln2b, h16);

  gemm_f16<EP_UP><<<dim3(kDFF / 64, kT / 128), 256, 0, stream>>>(
      h16, wu16, b_up, nullptr, nullptr, nullptr, nullptr, up16, nullptr, kDFF, kD);

  gemm_f16<EP_DOWN><<<dim3(kD / 64, kT / 128), 256, 0, stream>>>(
      up16, wd16, b_down, x1, nullptr, nullptr, nullptr, nullptr, (float*)d_out,
      kD, kDFF);
}
